// LSTMHarmonizer_62388694942348
// MI455X (gfx1250) — compile-verified, matrix operand order checked by reading, not run on hardware
//
#include <hip/hip_runtime.h>
#include <hip/hip_bf16.h>

// ---------------------------------------------------------------------------
// LSTM (B=64,T=1024,D=128,H=256) + 3 heads (V=128) for MI455X / gfx1250.
// bf16 WMMA (v_wmma_f32_16x16x32_bf16) with f32 accumulation.
//   K0  f32->bf16 conversion of x, W_ih, W_hh, head_w
//   K1  gx = x_bf @ W_ih_bf^T + (b_ih+b_hh)        (16x64 tile / wave)
//   K2  recurrent scan: 4 WGs x 16 batches; TDM double-buffers gx into LDS
//   K3  logits = hs_bf @ head_w_bf^T + head_b      (16x64 tile / wave)
// ---------------------------------------------------------------------------

#define B_  64
#define T_  1024
#define D_  128
#define H_  256
#define G_  1024          // 4*H
#define V_  128
#define NV_ 3
#define M_  (B_ * T_)     // 65536 rows

typedef __attribute__((ext_vector_type(16))) __bf16 v16bf;
typedef __attribute__((ext_vector_type(8)))  __bf16 v8bf;
typedef __attribute__((ext_vector_type(8)))  float  v8f;
typedef __attribute__((ext_vector_type(4)))  unsigned int v4u;
typedef __attribute__((ext_vector_type(8)))  int v8i;
typedef __attribute__((ext_vector_type(4)))  int v4i;

// --------------------------- WMMA helpers ----------------------------------

__device__ __forceinline__ v8f wmma_bf16(v16bf a, v16bf b, v8f c) {
  return __builtin_amdgcn_wmma_f32_16x16x32_bf16(
      false, a, false, b, (short)0, c, false, false);
}

// 16x32 bf16 fragment from row-major storage (leading dim ld, elements).
// ISA 7.12.2: lane = 16*khalf + m; per lane two contiguous 8-elem K runs.
// Same pattern serves B operands sourced from N-major ([N,K] row-major) W.
__device__ __forceinline__ v16bf load_frag16x32(const __bf16* __restrict__ p,
                                                int ld) {
  const int lane = threadIdx.x & 31;
  const int r    = lane & 15;
  const int kb   = (lane >> 4) << 3;       // 0 or 8
  const __bf16* row = p + (size_t)r * ld;
  v8bf lo = *reinterpret_cast<const v8bf*>(row + kb);
  v8bf hi = *reinterpret_cast<const v8bf*>(row + 16 + kb);
  v16bf f;
#pragma unroll
  for (int i = 0; i < 8; ++i) { f[i] = lo[i]; f[8 + i] = hi[i]; }
  return f;
}

// C/D 16x16 f32 tile: lane = 16*(m>=8)+n ; VGPR r holds row m=r+8*(lane>=16)
__device__ __forceinline__ void store_c16x16(float* __restrict__ p, int ld, v8f c) {
  const int lane  = threadIdx.x & 31;
  const int n     = lane & 15;
  const int mbase = (lane >> 4) << 3;
#pragma unroll
  for (int r = 0; r < 8; ++r) p[(size_t)(mbase + r) * ld + n] = c[r];
}

// --------------------------- scalar math -----------------------------------

__device__ __forceinline__ float fast_sigmoid(float x) {
  return 1.0f / (1.0f + __expf(-x));
}
__device__ __forceinline__ float fast_tanh(float x) {
  float e = __expf(-2.0f * x);
  return (1.0f - e) / (1.0f + e);
}

// --------------------------- K0: convert -----------------------------------

__global__ __launch_bounds__(256) void f32_to_bf16_kernel(
    const float* __restrict__ src, __bf16* __restrict__ dst, int n) {
  int i = blockIdx.x * 256 + threadIdx.x;
  int stride = gridDim.x * 256;
  for (; i < n; i += stride) dst[i] = (__bf16)src[i];
}

// --------------------------- K1: gx GEMM -----------------------------------
// gx[m,n] = x_bf[m,:] . W_ih_bf[n,:] + b_ih[n] + b_hh[n]
// M=65536, N=1024, K=128. 16x64 output tile per wave (A reused x4).

__global__ __launch_bounds__(256) void gx_gemm_kernel(
    const __bf16* __restrict__ xb, const __bf16* __restrict__ wih,
    const float* __restrict__ bih, const float* __restrict__ bhh,
    float* __restrict__ gx) {
  const int wid = blockIdx.x * 8 + (threadIdx.x >> 5);
  const int mt  = wid >> 4;             // 4096 M tiles
  const int nq  = wid & 15;             // 16 N quads (64 wide)
  const int m0  = mt << 4;
  const int n0  = nq << 6;
  const int lane = threadIdx.x & 31;
  const int n    = lane & 15;

  v8f acc[4];
#pragma unroll
  for (int q = 0; q < 4; ++q) {
    const float bias = bih[n0 + 16 * q + n] + bhh[n0 + 16 * q + n];
#pragma unroll
    for (int r = 0; r < 8; ++r) acc[q][r] = bias;
  }

#pragma unroll
  for (int kk = 0; kk < D_; kk += 32) {
    v16bf a = load_frag16x32(xb + (size_t)m0 * D_ + kk, D_);
#pragma unroll
    for (int q = 0; q < 4; ++q) {
      v16bf b = load_frag16x32(wih + (size_t)(n0 + 16 * q) * D_ + kk, D_);
      acc[q] = wmma_bf16(a, b, acc[q]);
    }
  }
#pragma unroll
  for (int q = 0; q < 4; ++q)
    store_c16x16(gx + (size_t)m0 * G_ + n0 + 16 * q, G_, acc[q]);
}

// --------------------------- K2: recurrent scan ----------------------------
// 4 workgroups x 16 batch rows; 16 waves, wave w owns H-slice j=[16w,16w+16)
// of ALL FOUR gates (wave-local c/h update). h lives in LDS (8 KB).
// TDM double-buffers the next timestep's gx tile (16 x 1024 f32, row stride
// T*4H) into LDS while the current step's WMMAs run (s_wait_tensorcnt sync).

#define SCAN_LDS_H    0                  // 16*256 bf16  =   8 KB
#define SCAN_LDS_GX   8192               // 2 * 64 KB    = 128 KB
#define SCAN_LDS_SIZE (8192 + 2 * 65536)

__device__ __forceinline__ void tdm_load_gx_tile(
    const float* __restrict__ gx_tile_base, char* smem, int buf) {
  // D# group 0: count=1 | lds_addr | 57-bit global_addr | type=2
  unsigned int lds_addr =
      (unsigned int)(size_t)(smem + SCAN_LDS_GX + buf * 65536);
  unsigned long long ga = (unsigned long long)(size_t)gx_tile_base;
  v4u g0;
  g0[0] = 1u;                                        // count=1, user mode
  g0[1] = lds_addr;                                  // LDS byte address
  g0[2] = (unsigned int)(ga & 0xFFFFFFFFu);          // global_addr[31:0]
  g0[3] = (unsigned int)((ga >> 32) & 0x1FFFFFFu)    // global_addr[56:32]
          | (2u << 30);                              // type=2 ("image")
  // D# group 1: data_size=4B; tensor 1024 x 16; tile 1024 x 16;
  // tensor_dim0_stride = T*G = 1048576 elements (batch-row stride).
  v8i g1;
  g1[0] = (2 << 16);                                 // data_size = 4 bytes
  g1[1] = (int)((G_ & 0xFFFF) << 16);                // tensor_dim0[15:0]
  g1[2] = (int)((G_ >> 16) | ((16 & 0xFFFF) << 16)); // dim0[31:16]|dim1[15:0]
  g1[3] = (int)((16 >> 16) | (G_ << 16));            // dim1[31:16]|tile_dim0
  g1[4] = 16;                                        // tile_dim1=16, dim2=0
  g1[5] = (int)(T_ * G_);                            // dim0_stride[31:0]
  g1[6] = 0;                                         // stride hi | dim1_stride
  g1[7] = 0;
  v4i z4 = {0, 0, 0, 0};                             // groups 2/3 unused (2D)
  v8i z8 = {0, 0, 0, 0, 0, 0, 0, 0};
  // 6-arg toolchain variant: (g0, g1, g2, g3, extra, cpol)
  __builtin_amdgcn_tensor_load_to_lds(g0, g1, z4, z4, z8, 0);
}

__global__ __launch_bounds__(512) void lstm_scan_kernel(
    const float* __restrict__ gx, const __bf16* __restrict__ whh,
    __bf16* __restrict__ hs) {
  extern __shared__ char smem[];
  __bf16* h_lds  = (__bf16*)(smem + SCAN_LDS_H);   // [16][256]
  float*  gx_lds = (float*)(smem + SCAN_LDS_GX);   // [2][16][1024]

  const int wave  = threadIdx.x >> 5;   // 0..15
  const int lane  = threadIdx.x & 31;
  const int n     = lane & 15;          // col within 16-wide j tile
  const int mbase = (lane >> 4) << 3;   // 0 or 8 (batch sub-rows)
  const int b0    = blockIdx.x << 4;    // batch slice base
  const int j0    = wave << 4;          // H slice base

  const float* gx_base = gx + (size_t)b0 * T_ * G_;  // row (b0, t=0)

  // h = 0, c = 0; prefetch gx tile for t=0
  for (int i = threadIdx.x; i < 16 * H_; i += 512)
    h_lds[i] = (__bf16)0.0f;
  v8f c_reg;
#pragma unroll
  for (int r = 0; r < 8; ++r) c_reg[r] = 0.0f;

  if (wave == 0) {
    tdm_load_gx_tile(gx_base, smem, 0);
    __builtin_amdgcn_s_wait_tensorcnt(0);
  }
  __syncthreads();

  for (int t = 0; t < T_; ++t) {
    // Kick DMA for the next step's gx tile into the other LDS buffer.
    if (wave == 0 && t + 1 < T_)
      tdm_load_gx_tile(gx_base + (size_t)(t + 1) * G_, smem, (t + 1) & 1);

    // Accumulators start at input contribution gx (from LDS staging buffer).
    const float* gxt = gx_lds + (size_t)(t & 1) * (16 * G_);
    v8f acc[4];
#pragma unroll
    for (int g = 0; g < 4; ++g)
#pragma unroll
      for (int r = 0; r < 8; ++r)
        acc[g][r] = gxt[(mbase + r) * G_ + g * H_ + j0 + n];

    // g += h @ W_hh^T : K=256, 8 WMMA k-steps x 4 gate tiles.
#pragma unroll
    for (int kk = 0; kk < H_; kk += 32) {
      v16bf a = load_frag16x32(&h_lds[kk], H_);       // LDS A operand
#pragma unroll
      for (int g = 0; g < 4; ++g) {
        v16bf b = load_frag16x32(
            whh + (size_t)(g * H_ + j0) * H_ + kk, H_);
        acc[g] = wmma_bf16(a, b, acc[g]);
      }
    }

    // Gate math (wave-local: same (b,j) owns i,f,g,o).
    v8f hv;
#pragma unroll
    for (int r = 0; r < 8; ++r) {
      float ig = fast_sigmoid(acc[0][r]);
      float fg = fast_sigmoid(acc[1][r]);
      float gg = fast_tanh(acc[2][r]);
      float og = fast_sigmoid(acc[3][r]);
      float c  = fg * c_reg[r] + ig * gg;
      c_reg[r] = c;
      hv[r]    = og * fast_tanh(c);
    }

    __syncthreads();                    // all reads of h_lds / gx buf done
#pragma unroll
    for (int r = 0; r < 8; ++r) {
      __bf16 hb = (__bf16)hv[r];
      h_lds[(mbase + r) * H_ + j0 + n] = hb;
      hs[((size_t)(b0 + mbase + r) * T_ + t) * H_ + j0 + n] = hb;
    }
    if (wave == 0 && t + 1 < T_)
      __builtin_amdgcn_s_wait_tensorcnt(0);  // next gx tile landed
    __syncthreads();                    // h_lds + gx buffer ready
  }
}

// --------------------------- K3: head GEMM ---------------------------------
// logits[nv][m,v] = hs_bf[m,:] . head_w_bf[nv,v,:] + head_b[nv,v]
// M=65536, N=3*128, K=256. 16x64 tile per wave; out = 3 concat [B,T,V] f32.

__global__ __launch_bounds__(256) void head_gemm_kernel(
    const __bf16* __restrict__ hsb, const __bf16* __restrict__ hw,
    const float* __restrict__ hb, float* __restrict__ out) {
  const int wid = blockIdx.x * 8 + (threadIdx.x >> 5);
  const int mt  = wid / 6;              // 4096 M tiles
  const int nq  = wid % 6;              // 6 N quads (24 tiles)
  const int m0  = mt << 4;
  const int lane = threadIdx.x & 31;
  const int n    = lane & 15;

  v8f acc[4];
#pragma unroll
  for (int q = 0; q < 4; ++q) {
    const int tile = nq * 4 + q;        // quads never straddle a head
    const int nv = tile >> 3, v0 = (tile & 7) << 4;
    const float bias = hb[nv * V_ + v0 + n];
#pragma unroll
    for (int r = 0; r < 8; ++r) acc[q][r] = bias;
  }

#pragma unroll
  for (int kk = 0; kk < H_; kk += 32) {
    v16bf a = load_frag16x32(hsb + (size_t)m0 * H_ + kk, H_);
#pragma unroll
    for (int q = 0; q < 4; ++q) {
      const int tile = nq * 4 + q;
      const int nv = tile >> 3, v0 = (tile & 7) << 4;
      v16bf b = load_frag16x32(hw + (size_t)(nv * V_ + v0) * H_ + kk, H_);
      acc[q] = wmma_bf16(a, b, acc[q]);
    }
  }
#pragma unroll
  for (int q = 0; q < 4; ++q) {
    const int tile = nq * 4 + q;
    const int nv = tile >> 3, v0 = (tile & 7) << 4;
    store_c16x16(out + (size_t)nv * M_ * V_ + (size_t)m0 * V_ + v0, V_, acc[q]);
  }
}

// --------------------------- launch ----------------------------------------

extern "C" void kernel_launch(void* const* d_in, const int* in_sizes, int n_in,
                              void* d_out, int out_size, void* d_ws, size_t ws_size,
                              hipStream_t stream) {
  const float* x      = (const float*)d_in[0];   // [B,T,D]
  const float* W_ih   = (const float*)d_in[1];   // [4H,D]
  const float* W_hh   = (const float*)d_in[2];   // [4H,H]
  const float* b_ih   = (const float*)d_in[3];   // [4H]
  const float* b_hh   = (const float*)d_in[4];   // [4H]
  const float* head_w = (const float*)d_in[5];   // [NV,V,H]
  const float* head_b = (const float*)d_in[6];   // [NV,V]
  float* out = (float*)d_out;                    // 3x[B,T,V] concat

  char* ws = (char*)d_ws;
  size_t off = 0;
  auto alloc = [&](size_t bytes) {
    void* p = ws + off;
    off = (off + bytes + 255) & ~(size_t)255;
    return p;
  };
  __bf16* xb   = (__bf16*)alloc((size_t)M_ * D_ * 2);        // 16 MB
  __bf16* wihb = (__bf16*)alloc((size_t)G_ * D_ * 2);        // 256 KB
  __bf16* whhb = (__bf16*)alloc((size_t)G_ * H_ * 2);        // 512 KB
  __bf16* hwb  = (__bf16*)alloc((size_t)NV_ * V_ * H_ * 2);  // 192 KB
  __bf16* hsb  = (__bf16*)alloc((size_t)M_ * H_ * 2);        // 32 MB
  float*  gx   = (float*)alloc((size_t)M_ * G_ * 4);         // 256 MB

  f32_to_bf16_kernel<<<2048, 256, 0, stream>>>(x,      xb,   M_ * D_);
  f32_to_bf16_kernel<<<512,  256, 0, stream>>>(W_ih,   wihb, G_ * D_);
  f32_to_bf16_kernel<<<1024, 256, 0, stream>>>(W_hh,   whhb, G_ * H_);
  f32_to_bf16_kernel<<<384,  256, 0, stream>>>(head_w, hwb,  NV_ * V_ * H_);

  // K1: 4096*16 wave jobs / 8 per WG
  gx_gemm_kernel<<<(4096 * 16) / 8, 256, 0, stream>>>(xb, wihb, b_ih, b_hh, gx);

  // K2: recurrence, dynamic LDS = 8 KB h + 2x64 KB gx staging
  lstm_scan_kernel<<<4, 512, SCAN_LDS_SIZE, stream>>>(gx, whhb, hsb);

  // K3: 4096*6 wave jobs / 8 per WG
  head_gemm_kernel<<<(4096 * 6) / 8, 256, 0, stream>>>(hsb, hwb, head_b, out);
}
